// GemNetOC_704374636661
// MI455X (gfx1250) — matrix operand order repaired and checked by reading, not compiled, not run on hardware
//
#include <hip/hip_runtime.h>

// ---------------------------------------------------------------------------
// CDNA5 (gfx1250) GemNet-OC forward.  wave32, WMMA bf16 -> f32 accumulate.
// Async GLOBAL->LDS double-buffered staging when the toolchain exposes it.
// ---------------------------------------------------------------------------

#if defined(__AMDGCN__) && __has_builtin(__builtin_amdgcn_global_load_async_to_lds_b128)
#define HAVE_ASYNC_LDS 1
#else
#define HAVE_ASYNC_LDS 0
#endif

typedef __attribute__((ext_vector_type(16))) __bf16 v16bf;
typedef __attribute__((ext_vector_type(8)))  float  v8f;
typedef __attribute__((ext_vector_type(4)))  float  f4;
typedef __attribute__((ext_vector_type(4)))  int    v4i;

union FragBF {
    struct { uint4 lo, hi; } q;   // 32 bytes
    v16bf v;                      // 16 x bf16
};

#define MODE_NONE            0
#define MODE_SILU            1
#define MODE_ADD_SILU_SCALE  2   // out = (Cin + silu(acc)) * scale
#define MODE_ACCUM           3   // out += acc

static __device__ __forceinline__ float siluf(float v) {
    return v / (1.0f + __expf(-v));
}

// fp32 -> bf16 RNE
static __device__ __forceinline__ unsigned short f2bf(float f) {
    unsigned int u = __float_as_uint(f);
    u += 0x7fffu + ((u >> 16) & 1u);
    return (unsigned short)(u >> 16);
}

// pack 4 fp32 -> 4 bf16 (little-endian element order)
static __device__ __forceinline__ uint2 pack4bf(f4 v) {
    uint2 r;
    r.x = (unsigned)f2bf(v.x) | ((unsigned)f2bf(v.y) << 16);
    r.y = (unsigned)f2bf(v.z) | ((unsigned)f2bf(v.w) << 16);
    return r;
}

#if HAVE_ASYNC_LDS
// 16-byte async copy global -> LDS (tracked by ASYNCcnt).
// Intrinsic pointee type is a 4 x i32 vector (per clang signature).
static __device__ __forceinline__ void async_cp16(const float* g, void* lds) {
    __builtin_amdgcn_global_load_async_to_lds_b128(
        (__attribute__((address_space(1))) v4i*)(void*)g,
        (__attribute__((address_space(3))) v4i*)lds, 0, 0);
}
#endif

static __device__ __forceinline__ void wait_async0() {
#if HAVE_ASYNC_LDS
#if __has_builtin(__builtin_amdgcn_s_wait_asynccnt)
    __builtin_amdgcn_s_wait_asynccnt(0);
#else
    asm volatile("s_wait_asynccnt 0x0" ::: "memory");
#endif
#endif
}

// ---------------------------------------------------------------------------
// Unified WMMA GEMM:  Out[M,Nc] = epilogue( A[M,K] @ Bw[K,Nc] )
//   CONCAT=0: A = Ap (row-major [M,K])
//   CONCAT=1: A[row,k] = k<w0 ? s0[i0[row],k] : k<w0+w1 ? s1[i1[row],k-w0]
//                              : Ap[row, k-w0-w1]   (row width w2 = K-w0-w1)
// Block: 256 threads (8 wave32), tile 128(M) x 64(N), K-step 32.
// All call sites use K % 32 == 0 and Nc % 64 == 0; concat segment widths are
// multiples of 32, so each K-chunk lies in exactly one segment (uniform).
// ---------------------------------------------------------------------------
template <int CONCAT>
__global__ __launch_bounds__(256)
void wmma_gemm_t(const float* __restrict__ Ap,
                 const float* __restrict__ s0, const int* __restrict__ i0, int w0,
                 const float* __restrict__ s1, const int* __restrict__ i1, int w1, int w2,
                 const float* __restrict__ Bw, const float* bias, const float* Cin,
                 float* Out, int M, int K, int Nc, int mode, float scale)
{
    __shared__ unsigned short As[128 * 32];   // bf16 tile [m][k], 8 KB
    __shared__ unsigned short Bs[64 * 32];    // bf16 tile [n][k] n-major, 4 KB
#if HAVE_ASYNC_LDS
    __shared__ float Af[2][128 * 32];         // fp32 async landing, 32 KB
    __shared__ float Bf[2][32 * 64];          // fp32 async landing, 16 KB
#endif

    const int tid  = threadIdx.x;
    const int lane = tid & 31;
    const int wave = tid >> 5;
    const int half = lane >> 4;
    const int l16  = lane & 15;
    const int m0   = blockIdx.x * 128;
    const int n0   = blockIdx.y * 64;

    v8f acc[4] = {};

    // base address of 32-wide A chunk for row gr at k-offset kb
    auto a_base = [&](int gr, int kb) -> const float* {
        if (CONCAT) {
            if (kb < w0)      return s0 + (long)i0[gr] * w0 + kb;
            if (kb < w0 + w1) return s1 + (long)i1[gr] * w1 + (kb - w0);
            return Ap + (long)gr * w2 + (kb - w0 - w1);
        }
        return Ap + (long)gr * K + kb;
    };

    auto compute_tile = [&]() {
        const uint4* arow = (const uint4*)&As[(wave * 16 + l16) * 32];
        FragBF fa;
        fa.q.lo = arow[half];        // K = half*8 .. +7
        fa.q.hi = arow[2 + half];    // K = 16+half*8 .. +7
        FragBF fb[4];
        #pragma unroll
        for (int nt = 0; nt < 4; ++nt) {
            const uint4* brow = (const uint4*)&Bs[(nt * 16 + l16) * 32];
            fb[nt].q.lo = brow[half];
            fb[nt].q.hi = brow[2 + half];
        }
        #pragma unroll
        for (int nt = 0; nt < 4; ++nt)
            acc[nt] = __builtin_amdgcn_wmma_f32_16x16x32_bf16(
                false, fa.v, false, fb[nt].v, (short)0, acc[nt], false, false);
    };

#if HAVE_ASYNC_LDS
    auto stage_async = [&](int buf, int kb) {
        #pragma unroll
        for (int i = 0; i < 4; ++i) {          // A: 128x32 fp32, float4/lane
            int lin = tid + 256 * i;           // 0..1023 float4 slots
            int r = lin >> 3, q = lin & 7;
            int gr = m0 + r;
            if (gr < M) async_cp16(a_base(gr, kb) + q * 4, &Af[buf][r * 32 + q * 4]);
        }
        #pragma unroll
        for (int i = 0; i < 2; ++i) {          // B: 32x64 fp32, k-major
            int lin = tid + 256 * i;           // 0..511 float4 slots
            int kr = lin >> 4, q = lin & 15;
            async_cp16(Bw + (long)(kb + kr) * Nc + n0 + q * 4,
                       &Bf[buf][kr * 64 + q * 4]);
        }
    };
    auto convert = [&](int buf) {
        #pragma unroll
        for (int i = 0; i < 4; ++i) {          // A fp32 -> bf16 (same layout)
            int lin = tid + 256 * i;
            f4 v = ((const f4*)Af[buf])[lin];
            ((uint2*)As)[lin] = pack4bf(v);
        }
        #pragma unroll
        for (int i = 0; i < 8; ++i) {          // B k-major fp32 -> n-major bf16
            int lin = tid + 256 * i;           // 0..2047
            int n = lin & 63, k = lin >> 6;
            Bs[n * 32 + k] = f2bf(Bf[buf][k * 64 + n]);
        }
    };
#else
    auto stage_sync = [&](int kb) {
        #pragma unroll
        for (int i = 0; i < 4; ++i) {          // A: vectorized b128 load + convert
            int lin = tid + 256 * i;
            int r = lin >> 3, q = lin & 7;
            int gr = m0 + r;
            if (gr < M) {
                f4 v = *(const f4*)(a_base(gr, kb) + q * 4);
                ((uint2*)As)[lin] = pack4bf(v);
            }
        }
        #pragma unroll
        for (int i = 0; i < 2; ++i) {          // B: b128 load, transpose to n-major
            int lin = tid + 256 * i;
            int kr = lin >> 4, q = lin & 15;
            f4 v = *(const f4*)(Bw + (long)(kb + kr) * Nc + n0 + q * 4);
            #pragma unroll
            for (int j = 0; j < 4; ++j)
                Bs[(q * 4 + j) * 32 + kr] = f2bf(v[j]);
        }
        if (kb + 32 < K)   // -> global_prefetch_b8
            __builtin_prefetch(Bw + (long)(kb + 32) * Nc + n0 + (tid & 63), 0, 1);
    };
#endif

    const int nkt = K >> 5;

#if HAVE_ASYNC_LDS
    stage_async(0, 0);
    for (int t = 0; t < nkt; ++t) {
        int buf = t & 1;
        wait_async0();                 // my async portions of buf complete
        __syncthreads();               // everyone's portions visible
        if (t + 1 < nkt) stage_async(buf ^ 1, (t + 1) * 32);   // overlap next tile
        convert(buf);
        __syncthreads();               // bf16 tiles ready
        compute_tile();
        __syncthreads();               // done reading As/Bs
    }
#else
    for (int t = 0; t < nkt; ++t) {
        stage_sync(t * 32);
        __syncthreads();
        compute_tile();
        __syncthreads();
    }
#endif

    // store (C/D layout: lanes 0-15 M=r, lanes 16-31 M=8+r; N = lane&15)
    #pragma unroll
    for (int nt = 0; nt < 4; ++nt) {
        #pragma unroll
        for (int r = 0; r < 8; ++r) {
            int gm = m0 + wave * 16 + half * 8 + r;
            int gn = n0 + nt * 16 + l16;
            if (gm < M) {
                long o = (long)gm * Nc + gn;
                float v = acc[nt][r];
                if (bias) v += bias[gn];
                float out;
                if (mode == MODE_SILU)                out = siluf(v);
                else if (mode == MODE_ADD_SILU_SCALE) out = (Cin[o] + siluf(v)) * scale;
                else if (mode == MODE_ACCUM)          out = Out[o] + v;
                else                                  out = v;
                Out[o] = out;
            }
        }
    }
}

// ---------------------------------------------------------------------------
// Elementwise / scatter kernels
// ---------------------------------------------------------------------------
__global__ void k_zero(float* p, long n) {
    long i = (long)blockIdx.x * 256 + threadIdx.x;
    if (i < n) p[i] = 0.0f;
}

__global__ void k_gather_emb(const int* an, const float* emb, float* h, long n) {
    long i = (long)blockIdx.x * 256 + threadIdx.x;
    if (i < n) h[i] = emb[(long)an[i >> 8] * 256 + (i & 255)];
}

__global__ void k_rbf(const float* D, float* rbf, long n) {
    long i = (long)blockIdx.x * 256 + threadIdx.x;
    if (i >= n) return;
    long e = i >> 7;
    int  r = (int)(i & 127);
    float d    = D[e] * (1.0f / 6.0f);          // d = D / CUTOFF
    float off  = (float)r * (1.0f / 127.0f);    // linspace(0,1,128)
    float coef = -0.5f * 127.0f * 127.0f;       // -0.5/step^2
    float diff = d - off;
    float g = __expf(coef * diff * diff);
    float d2 = d * d, d4 = d2 * d2, d5 = d4 * d;
    float env = 1.0f + d5 * (-21.0f + d * (35.0f - 15.0f * d));  // p=5 envelope
    if (d >= 1.0f) env = 0.0f;
    rbf[i] = g * env;
}

__global__ void k_cbf(const float* angle, float* cbf, long n) {
    long i = (long)blockIdx.x * 256 + threadIdx.x;
    if (i >= n) return;
    long t = i / 7;
    int  s = (int)(i - t * 7);
    cbf[i] = __cosf(angle[t] * (float)s);
}

__global__ void k_mul(const float* a, const float* b, float* c, long n) {
    long i = (long)blockIdx.x * 256 + threadIdx.x;
    if (i < n) c[i] = a[i] * b[i];
}

// dest[idx[row], col] += a[row,col] * g[row,col]
__global__ void k_mul_scatter(const float* a, const float* g, const int* idx,
                              float* dest, long M, int Nc) {
    long i = (long)blockIdx.x * 256 + threadIdx.x;
    if (i >= M * Nc) return;
    long row = i / Nc;
    int  col = (int)(i - row * Nc);
    atomicAdd(&dest[(long)idx[row] * Nc + col], a[i] * g[i]);
}

// agg[id3_ca[t], j] += xba[id3_ba[t], j] * dot(cbf[t,:], Wc[:,j])
__global__ void k_triplet(const float* xba, const float* cbf, const float* Wc,
                          const int* id3_ba, const int* id3_ca, float* agg, long T) {
    long i = (long)blockIdx.x * 256 + threadIdx.x;
    if (i >= T * 64) return;
    long t = i >> 6;
    int  j = (int)(i & 63);
    float dot = 0.0f;
    #pragma unroll
    for (int s = 0; s < 7; ++s) dot += cbf[t * 7 + s] * Wc[s * 64 + j];
    float v = xba[(long)id3_ba[t] * 64 + j] * dot;
    atomicAdd(&agg[(long)id3_ca[t] * 64 + j], v);
}

__global__ __launch_bounds__(256)
void k_energy(const float* Ea, const float* Wen, float* out, long n) {
    __shared__ float red[256];
    long i = (long)blockIdx.x * 256 + threadIdx.x;
    float v = 0.0f;
    if (i < n) v = siluf(Ea[i]) * Wen[(int)(i & 255)];
    red[threadIdx.x] = v;
    __syncthreads();
    for (int s = 128; s > 0; s >>= 1) {
        if (threadIdx.x < s) red[threadIdx.x] += red[threadIdx.x + s];
        __syncthreads();
    }
    if (threadIdx.x == 0) atomicAdd(out, red[0]);
}

// ---------------------------------------------------------------------------
// Host orchestration
// ---------------------------------------------------------------------------
extern "C" void kernel_launch(void* const* d_in, const int* in_sizes, int n_in,
                              void* d_out, int out_size, void* d_ws, size_t ws_size,
                              hipStream_t stream)
{
    (void)n_in; (void)out_size; (void)ws_size;
    const int N = in_sizes[0];
    const int E = in_sizes[1];
    const int T = in_sizes[4];
    const int A = 256, Ee = 512, Ti = 64, R = 128, S = 7, B = 4;
    const float INV_SQRT2 = 0.7071067811865476f;

    const int*   an       = (const int*)  d_in[0];
    const int*   idx_s    = (const int*)  d_in[1];
    const int*   idx_t    = (const int*)  d_in[2];
    const float* D_st     = (const float*)d_in[3];
    const float* angle3   = (const float*)d_in[4];
    const int*   id3_ba   = (const int*)  d_in[5];
    const int*   id3_ca   = (const int*)  d_in[6];
    const float* emb      = (const float*)d_in[7];
    const float* W_edge   = (const float*)d_in[8];
    const float* b_edge   = (const float*)d_in[9];
    const float* W_down   = (const float*)d_in[10];
    const float* W_rbf3   = (const float*)d_in[11];
    const float* W_cbf3   = (const float*)d_in[12];
    const float* W_up     = (const float*)d_in[13];
    const float* W_res1   = (const float*)d_in[14];
    const float* W_res2   = (const float*)d_in[15];
    const float* W_e2a    = (const float*)d_in[16];
    const float* W_rbf_h  = (const float*)d_in[17];
    const float* W_ares   = (const float*)d_in[18];
    const float* W_eemb   = (const float*)d_in[19];
    const float* W_rbf_out= (const float*)d_in[20];
    const float* W_out    = (const float*)d_in[21];
    const float* W_energy = (const float*)d_in[22];

    // workspace carve (fp32); every size except cbf's (T*S + 64) is a multiple of 64 floats
    float* p = (float*)d_ws;
    auto take = [&](long nf) { float* q = p; p += nf; return q; };
    float* h      = take((long)N * A);
    float* rbf    = take((long)E * R);
    float* x0     = take((long)E * Ee);
    float* x1     = take((long)E * Ee);
    float* gate_e = take((long)E * Ee);
    float* gate_a = take((long)E * A);
    float* gate_t = take((long)E * Ti);
    float* xba    = take((long)E * Ti);
    float* agg    = take((long)E * Ti);
    float* cbf    = take((long)T * S + 64);
    float* St     = take((long)N * Ee);
    float* amsg   = take((long)N * A);
    float* Ea     = take((long)N * A);

    auto blocks1d = [](long n) { return dim3((unsigned)((n + 255) / 256)); };
    auto zero = [&](float* q, long nf) {
        k_zero<<<blocks1d(nf), 256, 0, stream>>>(q, nf);
    };
    auto gemm = [&](const float* Aa, const float* Bb, const float* bias,
                    const float* Cin, float* Out, int M, int K, int Nc, int mode) {
        dim3 g((M + 127) / 128, Nc / 64);
        wmma_gemm_t<0><<<g, 256, 0, stream>>>(Aa, nullptr, nullptr, 0,
                                              nullptr, nullptr, 0, 0,
                                              Bb, bias, Cin, Out, M, K, Nc, mode, INV_SQRT2);
    };
    auto gemm_concat = [&](const float* s0, const int* i0, int w0,
                           const float* s1, const int* i1, int w1,
                           const float* s2, int w2,
                           const float* Bb, const float* bias, float* Out,
                           int M, int K, int Nc) {
        dim3 g((M + 127) / 128, Nc / 64);
        wmma_gemm_t<1><<<g, 256, 0, stream>>>(s2, s0, i0, w0, s1, i1, w1, w2,
                                              Bb, bias, nullptr, Out, M, K, Nc,
                                              MODE_SILU, 1.0f);
    };

    // --- precompute -------------------------------------------------------
    k_gather_emb<<<blocks1d((long)N * A), 256, 0, stream>>>(an, emb, h, (long)N * A);
    k_rbf<<<blocks1d((long)E * R), 256, 0, stream>>>(D_st, rbf, (long)E * R);
    k_cbf<<<blocks1d((long)T * S), 256, 0, stream>>>(angle3, cbf, (long)T * S);
    zero((float*)d_out, 1);

    float* cur = x0;
    float* oth = x1;

    // --- edge embedding: x = silu(concat(h_s, h_t, rbf) @ W_edge + b) -----
    gemm_concat(h, idx_s, A, h, idx_t, A, rbf, R, W_edge, b_edge, cur, E, 2 * A + R, Ee);

    // --- output block -----------------------------------------------------
    auto outblock = [&](int ob) {
        gemm(rbf, W_rbf_out + (long)ob * R * Ee, nullptr, nullptr, gate_e, E, R, Ee, MODE_NONE);
        zero(St, (long)N * Ee);
        k_mul_scatter<<<blocks1d((long)E * Ee), 256, 0, stream>>>(cur, gate_e, idx_t, St, E, Ee);
        gemm(St, W_out + (long)ob * Ee * A, nullptr, nullptr, Ea, N, Ee, A,
             ob == 0 ? MODE_NONE : MODE_ACCUM);
    };
    outblock(0);

    // --- interaction blocks ----------------------------------------------
    for (int b = 0; b < B; ++b) {
        // triplet path
        gemm(cur, W_down + (long)b * Ee * Ti, nullptr, nullptr, xba, E, Ee, Ti, MODE_SILU);
        gemm(rbf, W_rbf3 + (long)b * R * Ti, nullptr, nullptr, gate_t, E, R, Ti, MODE_NONE);
        k_mul<<<blocks1d((long)E * Ti), 256, 0, stream>>>(xba, gate_t, xba, (long)E * Ti);
        zero(agg, (long)E * Ti);
        k_triplet<<<blocks1d((long)T * 64), 256, 0, stream>>>(
            xba, cbf, W_cbf3 + (long)b * S * Ti, id3_ba, id3_ca, agg, T);
        gemm(agg, W_up + (long)b * Ti * Ee, nullptr, cur, cur, E, Ti, Ee, MODE_ADD_SILU_SCALE);

        // residual edge MLP
        gemm(cur, W_res1 + (long)b * Ee * Ee, nullptr, nullptr, oth, E, Ee, Ee, MODE_SILU);
        gemm(oth, W_res2 + (long)b * Ee * Ee, nullptr, cur, cur, E, Ee, Ee, MODE_ADD_SILU_SCALE);

        // atom update
        gemm(cur, W_e2a + (long)b * Ee * A, nullptr, nullptr, oth, E, Ee, A, MODE_NONE); // xe2a
        gemm(rbf, W_rbf_h + (long)b * R * A, nullptr, nullptr, gate_a, E, R, A, MODE_NONE);
        zero(amsg, (long)N * A);
        k_mul_scatter<<<blocks1d((long)E * A), 256, 0, stream>>>(oth, gate_a, idx_t, amsg, E, A);
        gemm(amsg, W_ares + (long)b * A * A, nullptr, h, h, N, A, A, MODE_ADD_SILU_SCALE);

        // re-embed edges from updated atoms
        gemm_concat(h, idx_s, A, h, idx_t, A, cur, Ee,
                    W_eemb + (long)b * (2 * A + Ee) * Ee, nullptr, oth, E, 2 * A + Ee, Ee);
        { float* t = cur; cur = oth; oth = t; }

        outblock(b + 1);
    }

    // --- energy -----------------------------------------------------------
    k_energy<<<blocks1d((long)N * A), 256, 0, stream>>>(Ea, W_energy, (float*)d_out, (long)N * A);
}
